// MultiTask_PLE_29205777613115
// MI455X (gfx1250) — matrix operand order repaired and checked by reading, not compiled, not run on hardware
//
#include <hip/hip_runtime.h>
#include <hip/hip_bf16.h>

// ---------------------------------------------------------------------------
// MultiTask PLE for MI455X (gfx1250).  All heavy GEMMs via v_wmma_f32_16x16x32_bf16.
// Weights pre-transposed to bf16 [N,K] so WMMA A/B fragments are aligned 16B loads
// matching the wave32 VGPR layouts (no LDS needed; weights are L2-resident).
// K-loop is a two-stage ping-pong pipeline (no register rotation copies):
// loads for stage s+1 issue before the WMMAs of stage s.
// ---------------------------------------------------------------------------

typedef __attribute__((ext_vector_type(16))) __bf16 v16bf;
typedef __attribute__((ext_vector_type(8)))  __bf16 bf16x8;
typedef __attribute__((ext_vector_type(8)))  float  v8f;

#define SHUF16(lo, hi) __builtin_shufflevector(lo, hi, 0,1,2,3,4,5,6,7,8,9,10,11,12,13,14,15)

static __device__ __forceinline__ v16bf make_frag(const __bf16* p, int hi_off) {
  bf16x8 lo = *(const bf16x8*)p;
  bf16x8 hi = *(const bf16x8*)(p + hi_off);
  return SHUF16(lo, hi);
}

// ---------------------------------------------------------------------------
// Generic bf16 WMMA GEMM:  out[z] = A[z][M,K] * W[z][N,K]^T (+bias), M=grid.x*128
// Wave tile: 16 rows x (NTILE*16) cols.  flags: 1=relu, 2=store bf16
// ---------------------------------------------------------------------------
template <int NTILE>
__global__ __launch_bounds__(256)
void gemm_bf16_wmma(const __bf16* __restrict__ A, long aStride,
                    const __bf16* __restrict__ W, long wStride,
                    const float* __restrict__ bias, long biasStride,
                    void* __restrict__ outp, long outStride,
                    int N, int K, int flags)
{
  const int lane = threadIdx.x & 31;
  const int wv   = threadIdx.x >> 5;
  const int lr   = lane & 15;
  const int lh   = lane >> 4;
  const int rowBase = blockIdx.x * 128 + wv * 16;
  const int colBase = blockIdx.y * (NTILE * 16);

  A += (long)blockIdx.z * aStride;
  W += (long)blockIdx.z * wStride;

  // A frag: row = rowBase+lr, K chunks {lh*8..+7, 16+lh*8..+7}
  const __bf16* Arow = A + (long)(rowBase + lr) * K + lh * 8;
  // B frag: col = colBase+j*16+lr, K chunks {lh*8..+7, 16+lh*8..+7}
  const __bf16* Wr[NTILE];
#pragma unroll
  for (int j = 0; j < NTILE; ++j)
    Wr[j] = W + (long)(colBase + j * 16 + lr) * K + lh * 8;

  v8f cc[NTILE];
#pragma unroll
  for (int j = 0; j < NTILE; ++j) cc[j] = (v8f){0.f,0.f,0.f,0.f,0.f,0.f,0.f,0.f};

  const int S = K >> 5;            // number of 32-wide K steps (1, 2, 8 or 16)

  // stage-0 prologue (k = 0)
  v16bf a0 = make_frag(Arow, 16);
  v16bf b0[NTILE];
#pragma unroll
  for (int j = 0; j < NTILE; ++j) b0[j] = make_frag(Wr[j], 16);

  v16bf a1;
  v16bf b1[NTILE];

  int s = 0;
  for (; s + 2 < S; s += 2) {
    // load stage 1 (step s+1)
    a1 = make_frag(Arow + (s + 1) * 32, 16);
#pragma unroll
    for (int j = 0; j < NTILE; ++j) b1[j] = make_frag(Wr[j] + (s + 1) * 32, 16);
    // compute stage 0 (step s)
#pragma unroll
    for (int j = 0; j < NTILE; ++j)
      cc[j] = __builtin_amdgcn_wmma_f32_16x16x32_bf16(false, a0, false, b0[j],
                                                      (short)0, cc[j], false, false);
    // load next stage 0 (step s+2)
    a0 = make_frag(Arow + (s + 2) * 32, 16);
#pragma unroll
    for (int j = 0; j < NTILE; ++j) b0[j] = make_frag(Wr[j] + (s + 2) * 32, 16);
    // compute stage 1 (step s+1)
#pragma unroll
    for (int j = 0; j < NTILE; ++j)
      cc[j] = __builtin_amdgcn_wmma_f32_16x16x32_bf16(false, a1, false, b1[j],
                                                      (short)0, cc[j], false, false);
  }

  if (S - s >= 2) {
    // final pair: steps s and s+1
    a1 = make_frag(Arow + (s + 1) * 32, 16);
#pragma unroll
    for (int j = 0; j < NTILE; ++j) b1[j] = make_frag(Wr[j] + (s + 1) * 32, 16);
#pragma unroll
    for (int j = 0; j < NTILE; ++j)
      cc[j] = __builtin_amdgcn_wmma_f32_16x16x32_bf16(false, a0, false, b0[j],
                                                      (short)0, cc[j], false, false);
#pragma unroll
    for (int j = 0; j < NTILE; ++j)
      cc[j] = __builtin_amdgcn_wmma_f32_16x16x32_bf16(false, a1, false, b1[j],
                                                      (short)0, cc[j], false, false);
  } else {
    // single remaining step (K == 32)
#pragma unroll
    for (int j = 0; j < NTILE; ++j)
      cc[j] = __builtin_amdgcn_wmma_f32_16x16x32_bf16(false, a0, false, b0[j],
                                                      (short)0, cc[j], false, false);
  }

  const float* bp = bias ? bias + (long)blockIdx.z * biasStride : nullptr;
  float*  outF = (float*)outp  + (long)blockIdx.z * outStride;
  __bf16* outB = (__bf16*)outp + (long)blockIdx.z * outStride;
  const int r0 = rowBase + lh * 8;

#pragma unroll
  for (int j = 0; j < NTILE; ++j) {
    const int col = colBase + j * 16 + lr;
    const float bv = bp ? bp[col] : 0.f;
#pragma unroll
    for (int v = 0; v < 8; ++v) {
      float r = cc[j][v] + bv;
      if (flags & 1) r = fmaxf(r, 0.f);
      long idx = (long)(r0 + v) * N + col;
      if (flags & 2) outB[idx] = (__bf16)r;
      else           outF[idx] = r;
    }
  }
}

// ---------------------------------------------------------------------------
// Prep kernels
// ---------------------------------------------------------------------------
__global__ void cvt_f32_to_bf16(const float* __restrict__ src, __bf16* __restrict__ dst, long n) {
  long i = (long)blockIdx.x * blockDim.x + threadIdx.x;
  if (i < n) dst[i] = (__bf16)src[i];
}

// src f32 [z][K,N] row-major -> dst bf16 [z][N,K] row-major
__global__ void transpose_to_bf16(const float* __restrict__ src, __bf16* __restrict__ dst, int K, int N) {
  src += (long)blockIdx.z * K * N;
  dst += (long)blockIdx.z * K * N;
  int i = blockIdx.x * blockDim.x + threadIdx.x;
  if (i < K * N) {
    int n = i % N, k = i / N;
    dst[(long)n * K + k] = (__bf16)src[i];
  }
}

// ---------------------------------------------------------------------------
// Row LayerNorm(64) + relu + cast to bf16
// ---------------------------------------------------------------------------
__global__ __launch_bounds__(256)
void ln_relu_to_bf16(const float* __restrict__ h, const float* __restrict__ g,
                     const float* __restrict__ beta, __bf16* __restrict__ out, int M)
{
  int r = blockIdx.x * blockDim.x + threadIdx.x;
  if (r >= M) return;
  const float4* p = (const float4*)(h + (long)r * 64);
  float v[64];
  float s = 0.f;
#pragma unroll
  for (int i = 0; i < 16; ++i) {
    float4 t = p[i];
    v[4*i+0] = t.x; v[4*i+1] = t.y; v[4*i+2] = t.z; v[4*i+3] = t.w;
    s += t.x + t.y + t.z + t.w;
  }
  float m = s * (1.f / 64.f);
  float vs = 0.f;
#pragma unroll
  for (int i = 0; i < 64; ++i) { float d = v[i] - m; vs += d * d; }
  float rstd = rsqrtf(vs * (1.f / 64.f) + 1e-5f);
  __bf16* o = out + (long)r * 64;
#pragma unroll
  for (int i = 0; i < 64; ++i) {
    float y = (v[i] - m) * rstd * g[i] + beta[i];
    o[i] = (__bf16)fmaxf(y, 0.f);
  }
}

// ---------------------------------------------------------------------------
// Gate: softmax(A[z] @ gW[z] + gb[z]) over E experts; thread per row.
// ---------------------------------------------------------------------------
template <typename TA, int E>
__global__ __launch_bounds__(256)
void gate_softmax(const TA* __restrict__ A, long aStride,
                  const float* __restrict__ gW, long gwStride,
                  const float* __restrict__ gb, long gbStride,
                  float* __restrict__ out, long outStride,
                  int Din, int M)
{
  int b = blockIdx.x * blockDim.x + threadIdx.x;
  if (b >= M) return;
  const TA* x = A + (long)blockIdx.z * aStride + (long)b * Din;
  const float* w  = gW + (long)blockIdx.z * gwStride;
  const float* bb = gb + (long)blockIdx.z * gbStride;
  float acc[E];
#pragma unroll
  for (int e = 0; e < E; ++e) acc[e] = bb[e];
  for (int d = 0; d < Din; ++d) {
    float xv = (float)x[d];
#pragma unroll
    for (int e = 0; e < E; ++e) acc[e] += xv * w[d * E + e];
  }
  float mx = acc[0];
#pragma unroll
  for (int e = 1; e < E; ++e) mx = fmaxf(mx, acc[e]);
  float s = 0.f;
#pragma unroll
  for (int e = 0; e < E; ++e) { acc[e] = __expf(acc[e] - mx); s += acc[e]; }
  float inv = 1.f / s;
  float* o = out + (long)blockIdx.z * outStride + (long)b * E;
#pragma unroll
  for (int e = 0; e < E; ++e) o[e] = acc[e] * inv;
}

// ---------------------------------------------------------------------------
// Gated combines.  Expert buffer layout: slot 0..8 = task (t*3+e), 9..12 = shared.
// ---------------------------------------------------------------------------
__global__ __launch_bounds__(256)
void combine_task(const float* __restrict__ gates, const float* __restrict__ Ebuf,
                  float* __restrict__ outF, __bf16* __restrict__ outB, int M)
{
  int idx = blockIdx.x * blockDim.x + threadIdx.x;
  if (idx >= 3 * M) return;
  int t = idx / M, b = idx - t * M;
  const float* gp = gates + (long)idx * 7;
  float g[7];
  const float* er[7];
#pragma unroll
  for (int e = 0; e < 7; ++e) {
    g[e] = gp[e];
    int slot = (e < 3) ? (t * 3 + e) : (6 + e);   // 3..6 -> shared 9..12
    er[e] = Ebuf + ((long)slot * M + b) * 32;
  }
  float* of = outF + (long)idx * 32;
  __bf16* ob = outB ? outB + (long)idx * 32 : nullptr;
#pragma unroll
  for (int o = 0; o < 32; ++o) {
    float acc = 0.f;
#pragma unroll
    for (int e = 0; e < 7; ++e) acc += g[e] * er[e][o];
    of[o] = acc;
    if (ob) ob[o] = (__bf16)acc;
  }
}

__global__ __launch_bounds__(256)
void combine_shared(const float* __restrict__ gates, const float* __restrict__ Ebuf,
                    float* __restrict__ outF, __bf16* __restrict__ outB, int M)
{
  int b = blockIdx.x * blockDim.x + threadIdx.x;
  if (b >= M) return;
  const float* gp = gates + (long)b * 13;
  float g[13];
  const float* er[13];
#pragma unroll
  for (int e = 0; e < 13; ++e) {
    g[e] = gp[e];
    er[e] = Ebuf + ((long)e * M + b) * 32;
  }
  float* of = outF + (long)b * 32;
  __bf16* ob = outB + (long)b * 32;
#pragma unroll
  for (int o = 0; o < 32; ++o) {
    float acc = 0.f;
#pragma unroll
    for (int e = 0; e < 13; ++e) acc += g[e] * er[e][o];
    of[o] = acc;
    ob[o] = (__bf16)acc;
  }
}

// ---------------------------------------------------------------------------
// Tower: out = relu(x @ W1[t] + b1[t]) @ W2[t] + b2[t]   (32 -> 32 -> 1)
// ---------------------------------------------------------------------------
__global__ __launch_bounds__(256)
void tower_kernel(const float* __restrict__ X, const float* __restrict__ W1,
                  const float* __restrict__ B1, const float* __restrict__ W2,
                  const float* __restrict__ B2, float* __restrict__ out, int M)
{
  int idx = blockIdx.x * blockDim.x + threadIdx.x;
  if (idx >= 3 * M) return;
  int t = idx / M;
  const float* x  = X + (long)idx * 32;
  const float* w1 = W1 + t * 1024;
  const float* b1 = B1 + t * 32;
  const float* w2 = W2 + t * 32;
  float xl[32];
#pragma unroll
  for (int i = 0; i < 32; ++i) xl[i] = x[i];
  float o = B2[t];
  for (int j = 0; j < 32; ++j) {
    float h = b1[j];
#pragma unroll
    for (int i = 0; i < 32; ++i) h += xl[i] * w1[i * 32 + j];
    o += fmaxf(h, 0.f) * w2[j];
  }
  out[idx] = o;
}

// ---------------------------------------------------------------------------
// Host orchestration
// ---------------------------------------------------------------------------
extern "C" void kernel_launch(void* const* d_in, const int* in_sizes, int n_in,
                              void* d_out, int out_size, void* d_ws, size_t ws_size,
                              hipStream_t stream)
{
  constexpr int   M = 32768;   // batch
  constexpr long  Ml = M;

  // JAX pytree flatten order (sorted dict keys): params{emb,layers,tow}, then x
  const float* embW1  = (const float*)d_in[0];   // [3,512,256]
  const float* embW2  = (const float*)d_in[1];   // [3,256,512]
  const float* embB1  = (const float*)d_in[2];   // [3,256]
  const float* embB2  = (const float*)d_in[3];   // [3,512]
  const float* l0gW   = (const float*)d_in[4];   // [3,512,7]
  const float* l0gb   = (const float*)d_in[5];   // [3,7]
  const float* l0sgW  = (const float*)d_in[6];   // [512,13]
  const float* l0sgb  = (const float*)d_in[7];   // [13]
  const float* l0shW1 = (const float*)d_in[8];   // [4,512,64]
  const float* l0shW2 = (const float*)d_in[9];   // [4,64,32]
  const float* l0shB1 = (const float*)d_in[10];  // [4,64]
  const float* l0shB2 = (const float*)d_in[11];  // [4,32]
  const float* l0shBe = (const float*)d_in[12];  // [4,64]
  const float* l0shG  = (const float*)d_in[13];  // [4,64]
  const float* l0tkW1 = (const float*)d_in[14];  // [3,3,512,64]
  const float* l0tkW2 = (const float*)d_in[15];  // [3,3,64,32]
  const float* l0tkB1 = (const float*)d_in[16];  // [3,3,64]
  const float* l0tkB2 = (const float*)d_in[17];  // [3,3,32]
  const float* l0tkBe = (const float*)d_in[18];  // [3,3,64]
  const float* l0tkG  = (const float*)d_in[19];  // [3,3,64]
  const float* l1gW   = (const float*)d_in[20];  // [3,32,7]
  const float* l1gb   = (const float*)d_in[21];  // [3,7]
  const float* l1shW1 = (const float*)d_in[22];  // [4,32,64]
  const float* l1shW2 = (const float*)d_in[23];  // [4,64,32]
  const float* l1shB1 = (const float*)d_in[24];
  const float* l1shB2 = (const float*)d_in[25];
  const float* l1shBe = (const float*)d_in[26];
  const float* l1shG  = (const float*)d_in[27];
  const float* l1tkW1 = (const float*)d_in[28];  // [3,3,32,64]
  const float* l1tkW2 = (const float*)d_in[29];  // [3,3,64,32]
  const float* l1tkB1 = (const float*)d_in[30];
  const float* l1tkB2 = (const float*)d_in[31];
  const float* l1tkBe = (const float*)d_in[32];
  const float* l1tkG  = (const float*)d_in[33];
  const float* towW1  = (const float*)d_in[34];  // [3,32,32]
  const float* towW2  = (const float*)d_in[35];  // [3,32,1]
  const float* towB1  = (const float*)d_in[36];  // [3,32]
  const float* towB2  = (const float*)d_in[37];  // [3,1]
  const float* x      = (const float*)d_in[38];  // [32768,512]

  // workspace carve-up
  char*  base = (char*)d_ws;
  size_t off  = 0;
  auto alloc = [&](size_t bytes) -> char* {
    char* p = base + off;
    off = (off + bytes + 255) & ~(size_t)255;
    return p;
  };
  __bf16* X_BF   = (__bf16*)alloc((size_t)M * 512 * 2);
  __bf16* EW1T   = (__bf16*)alloc((size_t)3 * 256 * 512 * 2);
  __bf16* EW2T   = (__bf16*)alloc((size_t)3 * 512 * 256 * 2);
  __bf16* L0W1T  = (__bf16*)alloc((size_t)13 * 64 * 512 * 2);
  __bf16* L0W2T  = (__bf16*)alloc((size_t)13 * 32 * 64 * 2);
  __bf16* L1W1T  = (__bf16*)alloc((size_t)13 * 64 * 32 * 2);
  __bf16* L1W2T  = (__bf16*)alloc((size_t)13 * 32 * 64 * 2);
  __bf16* H_BF   = (__bf16*)alloc((size_t)3 * M * 256 * 2);
  __bf16* TF_BF  = (__bf16*)alloc((size_t)3 * M * 512 * 2);
  float*  HEX_F  = (float*)alloc((size_t)M * 64 * 4);
  __bf16* HEX_BF = (__bf16*)alloc((size_t)M * 64 * 2);
  float*  E_F    = (float*)alloc((size_t)13 * M * 32 * 4);
  float*  GT_F   = (float*)alloc((size_t)3 * M * 7 * 4);
  float*  GS_F   = (float*)alloc((size_t)M * 13 * 4);
  float*  T0_F   = (float*)alloc((size_t)3 * M * 32 * 4);
  __bf16* T0_BF  = (__bf16*)alloc((size_t)3 * M * 32 * 2);
  float*  S0_F   = (float*)alloc((size_t)M * 32 * 4);
  __bf16* S0_BF  = (__bf16*)alloc((size_t)M * 32 * 2);
  float*  T1_F   = (float*)alloc((size_t)3 * M * 32 * 4);

  const dim3 blk(256);

  // --- prep: x -> bf16; weights -> bf16 transposed [N,K] ------------------
  {
    long n = (long)M * 512;
    cvt_f32_to_bf16<<<dim3((n + 255) / 256), blk, 0, stream>>>(x, X_BF, n);
  }
  auto tr = [&](const float* src, __bf16* dst, int K, int N, int z) {
    transpose_to_bf16<<<dim3((K * N + 255) / 256, 1, z), blk, 0, stream>>>(src, dst, K, N);
  };
  tr(embW1, EW1T, 512, 256, 3);
  tr(embW2, EW2T, 256, 512, 3);
  tr(l0tkW1, L0W1T,              512, 64, 9);
  tr(l0shW1, L0W1T + 9L*64*512,  512, 64, 4);
  tr(l0tkW2, L0W2T,               64, 32, 9);
  tr(l0shW2, L0W2T + 9L*32*64,    64, 32, 4);
  tr(l1tkW1, L1W1T,               32, 64, 9);
  tr(l1shW1, L1W1T + 9L*64*32,    32, 64, 4);
  tr(l1tkW2, L1W2T,               64, 32, 9);
  tr(l1shW2, L1W2T + 9L*32*64,    64, 32, 4);

  // --- task embedding: H = relu(x@W1+b1) bf16; TF = H@W2+b2 bf16 ----------
  gemm_bf16_wmma<4><<<dim3(M / 128, 256 / 64, 3), blk, 0, stream>>>(
      X_BF, 0L, EW1T, 256L * 512, embB1, 256L, H_BF, Ml * 256, 256, 512, /*relu|bf16*/3);
  gemm_bf16_wmma<4><<<dim3(M / 128, 512 / 64, 3), blk, 0, stream>>>(
      H_BF, Ml * 256, EW2T, 512L * 256, embB2, 512L, TF_BF, Ml * 512, 512, 256, /*bf16*/2);

  // --- layer-0 gates ------------------------------------------------------
  gate_softmax<__bf16, 7><<<dim3((M + 255) / 256, 1, 3), blk, 0, stream>>>(
      TF_BF, Ml * 512, l0gW, 512L * 7, l0gb, 7L, GT_F, Ml * 7, 512, M);
  gate_softmax<float, 13><<<dim3((M + 255) / 256, 1, 1), blk, 0, stream>>>(
      x, 0L, l0sgW, 0L, l0sgb, 0L, GS_F, 0L, 512, M);

  // --- layer-0 experts (slots 0..8 task, 9..12 shared) --------------------
  for (int slot = 0; slot < 13; ++slot) {
    const bool task = slot < 9;
    const __bf16* Ap = task ? TF_BF + (long)(slot / 3) * M * 512 : X_BF;
    const float* b1p = task ? l0tkB1 + slot * 64 : l0shB1 + (slot - 9) * 64;
    const float* b2p = task ? l0tkB2 + slot * 32 : l0shB2 + (slot - 9) * 32;
    const float* gp  = task ? l0tkG  + slot * 64 : l0shG  + (slot - 9) * 64;
    const float* bep = task ? l0tkBe + slot * 64 : l0shBe + (slot - 9) * 64;
    gemm_bf16_wmma<4><<<dim3(M / 128, 1, 1), blk, 0, stream>>>(
        Ap, 0L, L0W1T + (long)slot * 64 * 512, 0L, b1p, 0L, HEX_F, 0L, 64, 512, 0);
    ln_relu_to_bf16<<<dim3(M / 256), blk, 0, stream>>>(HEX_F, gp, bep, HEX_BF, M);
    gemm_bf16_wmma<2><<<dim3(M / 128, 1, 1), blk, 0, stream>>>(
        HEX_BF, 0L, L0W2T + (long)slot * 32 * 64, 0L, b2p, 0L,
        E_F + (long)slot * M * 32, 0L, 32, 64, 0);
  }

  // --- layer-0 combines ---------------------------------------------------
  combine_task<<<dim3((3 * M + 255) / 256), blk, 0, stream>>>(GT_F, E_F, T0_F, T0_BF, M);
  combine_shared<<<dim3((M + 255) / 256), blk, 0, stream>>>(GS_F, E_F, S0_F, S0_BF, M);

  // --- layer-1 gates ------------------------------------------------------
  gate_softmax<float, 7><<<dim3((M + 255) / 256, 1, 3), blk, 0, stream>>>(
      T0_F, Ml * 32, l1gW, 32L * 7, l1gb, 7L, GT_F, Ml * 7, 32, M);

  // --- layer-1 experts ----------------------------------------------------
  for (int slot = 0; slot < 13; ++slot) {
    const bool task = slot < 9;
    const __bf16* Ap = task ? T0_BF + (long)(slot / 3) * M * 32 : S0_BF;
    const float* b1p = task ? l1tkB1 + slot * 64 : l1shB1 + (slot - 9) * 64;
    const float* b2p = task ? l1tkB2 + slot * 32 : l1shB2 + (slot - 9) * 32;
    const float* gp  = task ? l1tkG  + slot * 64 : l1shG  + (slot - 9) * 64;
    const float* bep = task ? l1tkBe + slot * 64 : l1shBe + (slot - 9) * 64;
    gemm_bf16_wmma<4><<<dim3(M / 128, 1, 1), blk, 0, stream>>>(
        Ap, 0L, L1W1T + (long)slot * 64 * 32, 0L, b1p, 0L, HEX_F, 0L, 64, 32, 0);
    ln_relu_to_bf16<<<dim3(M / 256), blk, 0, stream>>>(HEX_F, gp, bep, HEX_BF, M);
    gemm_bf16_wmma<2><<<dim3(M / 128, 1, 1), blk, 0, stream>>>(
        HEX_BF, 0L, L1W2T + (long)slot * 32 * 64, 0L, b2p, 0L,
        E_F + (long)slot * M * 32, 0L, 32, 64, 0);
  }

  // --- layer-1 combine (last CGC: task outputs only) ----------------------
  combine_task<<<dim3((3 * M + 255) / 256), blk, 0, stream>>>(GT_F, E_F, T1_F, nullptr, M);

  // --- towers -> d_out [3,B,1] -------------------------------------------
  tower_kernel<<<dim3((3 * M + 255) / 256), blk, 0, stream>>>(
      T1_F, towW1, towB1, towW2, towB2, (float*)d_out, M);

  (void)in_sizes; (void)n_in; (void)out_size; (void)ws_size;
}
